// Module_1_35433480192344
// MI455X (gfx1250) — compile-verified, matrix operand order checked by reading, not run on hardware
//
#include <hip/hip_runtime.h>

typedef __attribute__((ext_vector_type(16))) __bf16 v16bf;
typedef __attribute__((ext_vector_type(8)))  __bf16 v8bf;
typedef __attribute__((ext_vector_type(8)))  float  v8f;

#define NN   116      // graph size / C_IN
#define HH   128      // hidden dim
#define PAD  128      // padded dim (8 tiles of 16)
#define LDM  136      // LDS stride in bf16 elements (16B-aligned rows, bank-spread)
#define SCRLD 129     // f32 scratch stride (bank-spread)
#define MATB (PAD * LDM * 2)            // one bf16 matrix = 34816 bytes
#define SMEM_BYTES (4 * MATB + 512)     // 139776 bytes dynamic LDS

// Load a 16x32 bf16 fragment in the CDNA5 WMMA A/B register layout.
// For A-operand: sm is row-major, major = tile row. For B-operand: sm holds
// the matrix column-major (i.e. B^T row-major), major = tile col.
// Layout (ISA 7.12.2): lanes 0-15 carry K=[kb..kb+7] in v0..3 and K=[kb+16..kb+23]
// in v4..7; lanes 16-31 the same shifted by 8. Two aligned 16B LDS loads per lane.
__device__ __forceinline__ v16bf load_frag(const __bf16* sm, int major, int kbase) {
  const int lane = threadIdx.x & 31;
  const int mj = lane & 15;
  const int kh = (lane >> 4) << 3;          // 0 or 8
  const __bf16* p = sm + (major * 16 + mj) * LDM + kbase + kh;
  v8bf lo = *(const v8bf*)(p);
  v8bf hi = *(const v8bf*)(p + 16);
  return __builtin_shufflevector(lo, hi, 0,1,2,3,4,5,6,7,8,9,10,11,12,13,14,15);
}

// One 16-row output strip (tile row = wave) of a 128x128 GEMM, B operand
// software-pipelined one column-tile ahead so ds_load latency overlaps WMMA.
// MODE 0: store bf16 col-major to smOut (contiguous v8bf per lane)
// MODE 1: relu(x + bias) -> bf16 row-major scatter to smOut
// MODE 2: relu(x + bias) -> fp32 global store (rows < NN only)
template <int MODE>
__device__ __forceinline__ void gemm_strip(const __bf16* __restrict__ smA_,
                                           const __bf16* __restrict__ smB_,
                                           __bf16* __restrict__ smOut,
                                           float* __restrict__ gOut,
                                           const float* __restrict__ gBias) {
  const int tid   = threadIdx.x;
  const int lane  = tid & 31;
  const int wave  = tid >> 5;
  const int n0    = lane & 15;
  const int mbase = (lane >> 4) << 3;

  v16bf af[4];
#pragma unroll
  for (int kt = 0; kt < 4; ++kt) af[kt] = load_frag(smA_, wave, kt * 32);

  float bias[8];
  if (MODE != 0) {
#pragma unroll
    for (int tc = 0; tc < 8; ++tc) bias[tc] = gBias[tc * 16 + n0];
  }

  v16bf bcur[4], bnext[4];
#pragma unroll
  for (int kt = 0; kt < 4; ++kt) bcur[kt] = load_frag(smB_, 0, kt * 32);

#pragma unroll
  for (int tc = 0; tc < 8; ++tc) {
    if (tc < 7) {
#pragma unroll
      for (int kt = 0; kt < 4; ++kt) bnext[kt] = load_frag(smB_, tc + 1, kt * 32);
    }
    v8f c = {};
#pragma unroll
    for (int kt = 0; kt < 4; ++kt)
      c = __builtin_amdgcn_wmma_f32_16x16x32_bf16(false, af[kt], false, bcur[kt],
                                                  (short)0, c, false, false);
    if (MODE == 0) {
      v8bf o;
#pragma unroll
      for (int r = 0; r < 8; ++r) o[r] = (__bf16)c[r];
      *(v8bf*)(smOut + (tc * 16 + n0) * LDM + wave * 16 + mbase) = o;
    } else if (MODE == 1) {
#pragma unroll
      for (int r = 0; r < 8; ++r) {
        float v = fmaxf(c[r] + bias[tc], 0.f);
        smOut[(wave * 16 + mbase + r) * LDM + tc * 16 + n0] = (__bf16)v;
      }
    } else {
#pragma unroll
      for (int r = 0; r < 8; ++r) {
        int row = wave * 16 + mbase + r;
        if (row < NN) gOut[row * HH + tc * 16 + n0] = fmaxf(c[r] + bias[tc], 0.f);
      }
    }
#pragma unroll
    for (int kt = 0; kt < 4; ++kt) bcur[kt] = bnext[kt];  // SSA rename after unroll
  }
}

__global__ void __launch_bounds__(256)
gcn2_kernel(const float* __restrict__ a, const float* __restrict__ f,
            const float* __restrict__ W1, const float* __restrict__ b1,
            const float* __restrict__ W2, const float* __restrict__ b2,
            float* __restrict__ out) {
  extern __shared__ char smem[];
  __bf16* smA = (__bf16*)(smem);             // A_norm, row-major bf16
  __bf16* smW = (__bf16*)(smem + 1 * MATB);  // weights stored transposed (col-major)
  __bf16* smX = (__bf16*)(smem + 2 * MATB);  // f then h (row-major, A-operand)
  __bf16* smY = (__bf16*)(smem + 3 * MATB);  // X1 then X2 (col-major, B-operand)
  float*  scr  = (float*)(smem + 2 * MATB);  // f32 scratch, aliases smX+smY
  float*  rowd = (float*)(smem + 4 * MATB);  // 128 floats: deg^{-1/2}

  const int b   = blockIdx.x;
  const int tid = threadIdx.x;

  const float* ab = a + (size_t)b * NN * NN;
  const float* fb = f + (size_t)b * NN * NN;
  float*       ob = out + (size_t)b * NN * HH;

  // ---- Phase 1: A_norm = D^{-1/2} (|a| + I) D^{-1/2} in bf16 -------------
  for (int idx = tid; idx < NN * NN; idx += 256) {
    int r = idx / NN, c = idx - r * NN;
    float v = fabsf(ab[idx]);
    if (r == c) v += 1.0f;
    scr[r * SCRLD + c] = v;
  }
  __syncthreads();
  if (tid < NN) {
    float s = 0.f;
    for (int c = 0; c < NN; ++c) s += scr[tid * SCRLD + c];
    rowd[tid] = rsqrtf(s);
  } else if (tid < PAD) {
    rowd[tid] = 0.f;
  }
  __syncthreads();
  for (int idx = tid; idx < PAD * PAD; idx += 256) {
    int r = idx >> 7, c = idx & 127;
    float v = (r < NN && c < NN) ? scr[r * SCRLD + c] * rowd[r] * rowd[c] : 0.f;
    smA[r * LDM + c] = (__bf16)v;
  }
  __syncthreads();  // scratch dead

  // ---- Phase 2: load f (row-major) and W1^T (col-major) as bf16 ----------
  for (int idx = tid; idx < PAD * PAD; idx += 256) {
    int r = idx >> 7, c = idx & 127;
    float v = (r < NN && c < NN) ? fb[r * NN + c] : 0.f;
    smX[r * LDM + c] = (__bf16)v;
  }
  for (int idx = tid; idx < PAD * PAD; idx += 256) {
    int k = idx >> 7, n = idx & 127;           // W1[k][n], coalesced global read
    float v = (k < NN) ? W1[idx] : 0.f;
    smW[n * LDM + k] = (__bf16)v;              // store transposed
  }
  __syncthreads();

  // ---- Stage 1: X1 = f @ W1 -> smY (col-major) ---------------------------
  gemm_strip<0>(smX, smW, smY, nullptr, nullptr);
  __syncthreads();

  // ---- Stage 2: h = relu(A_norm @ X1 + b1) -> smX (row-major) ------------
  gemm_strip<1>(smA, smY, smX, nullptr, b1);
  // W2^T -> smW (W1 dead since last barrier); stage 2 never touches smW
  for (int idx = tid; idx < PAD * PAD; idx += 256) {
    int k = idx >> 7, n = idx & 127;
    smW[n * LDM + k] = (__bf16)W2[idx];
  }
  __syncthreads();

  // ---- Stage 3: X2 = h @ W2 -> smY (col-major) ---------------------------
  gemm_strip<0>(smX, smW, smY, nullptr, nullptr);
  __syncthreads();

  // ---- Stage 4: out = relu(A_norm @ X2 + b2), fp32 global stores ---------
  gemm_strip<2>(smA, smY, nullptr, ob, b2);
}

extern "C" void kernel_launch(void* const* d_in, const int* in_sizes, int n_in,
                              void* d_out, int out_size, void* d_ws, size_t ws_size,
                              hipStream_t stream) {
  (void)n_in; (void)out_size; (void)d_ws; (void)ws_size;
  const float* a  = (const float*)d_in[0];
  const float* f  = (const float*)d_in[1];
  const float* W1 = (const float*)d_in[2];
  const float* b1 = (const float*)d_in[3];
  const float* W2 = (const float*)d_in[4];
  const float* b2 = (const float*)d_in[5];
  float* out = (float*)d_out;

  const int B = in_sizes[0] / (NN * NN);  // 4096
  gcn2_kernel<<<dim3(B), dim3(256), SMEM_BYTES, stream>>>(a, f, W1, b1, W2, b2, out);
}
